// WaveFlow_70342974374555
// MI455X (gfx1250) — matrix operand order repaired and checked by reading, not compiled, not run on hardware
//
#include <hip/hip_runtime.h>
#include <hip/hip_bf16.h>
#include <cstdint>

typedef __attribute__((ext_vector_type(16))) __bf16 v16bf;
typedef __attribute__((ext_vector_type(8)))  __bf16 v8bf;
typedef __attribute__((ext_vector_type(8)))  float  v8f;
typedef __attribute__((ext_vector_type(4)))  int    v4i;

struct HPerm { int h[16]; };

#define WTOT 4000
#define HTOT 16
#define BTOT 2
#define NPIX (BTOT*HTOT*WTOT) /* 128000 */
#define NWTILE 63              /* ceil(4000/64) */

// padded LDS row strides (bank-conflict-free for 16-lane fragment reads)
#define BCS 72    /* res halo row stride (bf16)  : 144B = 36 dwords  */
#define CCS 104   /* cdt row stride (bf16)       : 208B = 52 dwords  */
#define GS  65    /* gate row stride (f32)       : 65 dwords         */
#define AS_ 72    /* act/skp/h1 row stride (bf16): 144B = 36 dwords  */

__device__ inline v8f v8f_zero(){ v8f z; for(int i=0;i<8;i++) z[i]=0.f; return z; }

// A fragment: lane holds 16 contiguous K values {c..c+15} (lanes 0-15) / {c+16..c+31} (16-31)
__device__ inline v16bf ldA16(const __bf16* p){
  v8bf lo = *(const v8bf*)p;
  v8bf hh = *(const v8bf*)(p+8);
  v16bf r;
  #pragma unroll
  for(int i=0;i<8;i++){ r[i]=lo[i]; r[i+8]=hh[i]; }
  return r;
}
// B fragment: lane holds 16 contiguous K values; built from 2x16B (16B-aligned ok)
__device__ inline v16bf ldC16(const __bf16* p){
  v8bf lo = *(const v8bf*)p;
  v8bf hh = *(const v8bf*)(p+8);
  v16bf r;
  #pragma unroll
  for(int i=0;i<8;i++){ r[i]=lo[i]; r[i+8]=hh[i]; }
  return r;
}

__device__ inline v8f wmma_bf(v16bf a, v16bf b, v8f c){
  // D = A(16x32 bf16) * B(32x16 bf16) + C(16x16 f32)
  return __builtin_amdgcn_wmma_f32_16x16x32_bf16(false, a, false, b, (short)0, c, false, false);
}

// ---- CDNA5 async global->LDS staging (guarded; falls back to VGPR round-trip)
#if defined(__HIP_DEVICE_COMPILE__) && __has_builtin(__builtin_amdgcn_global_load_async_to_lds_b128)
#define USE_ASYNC_LDS 1
typedef __attribute__((address_space(1))) v4i* gas1p;
typedef __attribute__((address_space(3))) v4i* las3p;
__device__ inline void cp16_async(const void* g, void* l){
  __builtin_amdgcn_global_load_async_to_lds_b128(
      (gas1p)(uintptr_t)g, (las3p)(uint32_t)(uintptr_t)l, 0, 0);
}
#else
#define USE_ASYNC_LDS 0
__device__ inline void cp16_async(const void* g, void* l){
  *(v8bf*)l = *(const v8bf*)g;
}
#endif
__device__ inline void zero16_lds(void* l){
  v8bf z; for(int t=0;t<8;t++) z[t] = (__bf16)0.f;
  *(v8bf*)l = z;
}
__device__ inline void wait_async(){
#if USE_ASYNC_LDS
#if __has_builtin(__builtin_amdgcn_s_wait_asynccnt)
  __builtin_amdgcn_s_wait_asynccnt(0);
#else
  asm volatile("s_wait_asynccnt 0x0" ::: "memory");
#endif
#endif
}

// ---------------------------------------------------------------- prep kernels
__global__ void k_cvt(const float* __restrict__ s, __bf16* __restrict__ d, int n){
  int i = blockIdx.x*256 + threadIdx.x;
  if(i < n) d[i] = (__bf16)s[i];
}
__global__ void k_cpy(const float* __restrict__ s, float* __restrict__ d, int n){
  int i = blockIdx.x*256 + threadIdx.x;
  if(i < n) d[i] = s[i];
}
// init_w [fl][o=128][i=64][kh][kw] -> [fl][o][t=kh*3+kw][i]
__global__ void k_prep_wconv(const float* __restrict__ s, __bf16* __restrict__ d){
  int idx = blockIdx.x*256 + threadIdx.x;
  if(idx >= 64*128*576) return;
  int i  = idx % 64;
  int t  = (idx/64) % 9;
  int o  = (idx/576) % 128;
  int fl = idx / (576*128);
  int kh = t/3, kw = t%3;
  d[idx] = (__bf16)s[ ((size_t)(fl*128 + o)*64 + i)*9 + kh*3 + kw ];
}
// cdt_w [fl][128][80] -> [fl][128][96] (zero pad)
__global__ void k_prep_wcdt(const float* __restrict__ s, __bf16* __restrict__ d){
  int idx = blockIdx.x*256 + threadIdx.x;
  if(idx >= 64*128*96) return;
  int i  = idx % 96;
  int o  = (idx/96) % 128;
  int fl = idx / (96*128);
  d[idx] = (i < 80) ? (__bf16)s[(size_t)(fl*128+o)*80 + i] : (__bf16)0.f;
}
// post2_w [f][2][64] -> [f][16][64] (zero pad rows)
__global__ void k_prep_wp2(const float* __restrict__ s, __bf16* __restrict__ d){
  int idx = blockIdx.x*256 + threadIdx.x;
  if(idx >= 8*16*64) return;
  int i = idx % 64;
  int o = (idx/64) % 16;
  int f = idx / (64*16);
  d[idx] = (o < 2) ? (__bf16)s[(size_t)(f*2+o)*64 + i] : (__bf16)0.f;
}
// c [b][80][h][w] f32 -> [b][h][w][96] bf16 (canonical orientation)
__global__ void k_prep_c(const float* __restrict__ c, __bf16* __restrict__ d){
  int pix = blockIdx.x*256 + threadIdx.x;
  if(pix >= NPIX) return;
  int w = pix % WTOT;
  int h = (pix / WTOT) % HTOT;
  int b = pix / (HTOT*WTOT);
  __bf16* o = d + (size_t)pix*96;
  for(int ch=0; ch<96; ch++)
    o[ch] = (ch < 80) ? (__bf16)c[ ((size_t)(b*80+ch)*HTOT + h)*WTOT + w ] : (__bf16)0.f;
}

// ---------------------------------------------------------------- first conv
__global__ void k_first(const float* __restrict__ Xc, const float* __restrict__ fw,
                        __bf16* __restrict__ res, HPerm P){
  int pix = blockIdx.x*256 + threadIdx.x;
  if(pix >= NPIX) return;
  int w = pix % WTOT;
  int h = (pix / WTOT) % HTOT;
  int b = pix / (HTOT*WTOT);
  float x0 = (h >= 2) ? Xc[(size_t)(b*HTOT + P.h[h-2])*WTOT + w] : 0.f;
  float x1 = (h >= 1) ? Xc[(size_t)(b*HTOT + P.h[h-1])*WTOT + w] : 0.f;
  __bf16* o = res + (size_t)pix*64;
  for(int k=0;k<64;k++) o[k] = (__bf16)tanhf(fw[k*2]*x0 + fw[k*2+1]*x1);
}

// ---------------------------------------------------------------- WaveNet layer (fused, WMMA)
__global__ __launch_bounds__(256) void k_layer(
    const __bf16* __restrict__ resIn, __bf16* __restrict__ resOut,
    __bf16* __restrict__ skp, const __bf16* __restrict__ cB,
    const __bf16* __restrict__ Wc,  const __bf16* __restrict__ Wq,
    const __bf16* __restrict__ Wr,  const __bf16* __restrict__ Wsk,
    int d, int firstLayer, HPerm P)
{
  __shared__ __align__(32) char smem[47872];
  __bf16* Bc  = (__bf16*)smem;            // phase1: res halo [3*80][BCS] bf16 (34560 B)
  __bf16* Cc  = (__bf16*)(smem + 34560);  // phase1: c tile   [64][CCS]  bf16 (13312 B)
  float*  G   = (float*)smem;             // phase2: gate pre-act [128][GS] f32 (33280 B)
  __bf16* ACT = (__bf16*)(smem + 33280);  // phase2: gated act [64][AS_] bf16  ( 9216 B)

  int bz = blockIdx.z, h = blockIdx.y, w0 = blockIdx.x*64;
  int tid = threadIdx.x, lane = tid & 31, wv = tid >> 5;
  int l = lane & 15, hi = lane >> 4;

  // ---- stage res halo: rows {h-2d,h-d,h}, cols [w0-d, w0+63+d]
  for(int idx = tid; idx < 240*8; idx += 256){
    int chunk = idx & 7;           // 8 bf16 = 16B
    int pos   = idx >> 3;          // (kh,col)
    int col   = pos % 80;
    int kh    = pos / 80;
    int gr    = h - (2-kh)*d;
    int gc    = w0 - d + col;
    void* dst = Bc + (size_t)pos*BCS + chunk*8;
    if(gr >= 0 && gc >= 0 && gc < WTOT)
      cp16_async(resIn + (((size_t)(bz*HTOT+gr)*WTOT + gc)*64 + chunk*8), dst);
    else
      zero16_lds(dst);
  }
  // ---- stage conditioner tile (perm'd row)
  {
    int hc = P.h[h];
    for(int idx = tid; idx < 64*12; idx += 256){
      int chunk = idx % 12;
      int col   = idx / 12;
      int gc    = w0 + col;
      void* dst = Cc + (size_t)col*CCS + chunk*8;
      if(gc < WTOT)
        cp16_async(cB + (((size_t)(bz*HTOT+hc)*WTOT + gc)*96 + chunk*8), dst);
      else
        zero16_lds(dst);
    }
  }
  wait_async();
  __syncthreads();

  // ---- g = dilated conv + cdt : wave wv owns out-channels [wv*16, wv*16+16)
  v8f acc[4]; for(int s=0;s<4;s++) acc[s] = v8f_zero();
  int row = wv*16 + l;
  #pragma unroll
  for(int t=0;t<9;t++){
    const int kh = t/3, kw = t%3;
    int cshift = kw*d;
    #pragma unroll
    for(int kc=0;kc<2;kc++){
      v16bf a = ldA16(Wc + (size_t)row*576 + t*64 + kc*32 + hi*16);
      #pragma unroll
      for(int s=0;s<4;s++){
        int col = s*16 + l + cshift;
        v16bf bb = ldC16(Bc + ((size_t)kh*80 + col)*BCS + kc*32 + hi*16);
        acc[s] = wmma_bf(a, bb, acc[s]);
      }
    }
  }
  #pragma unroll
  for(int kc=0;kc<3;kc++){
    v16bf a = ldA16(Wq + (size_t)row*96 + kc*32 + hi*16);
    #pragma unroll
    for(int s=0;s<4;s++){
      v16bf bb = ldC16(Cc + (size_t)(s*16+l)*CCS + kc*32 + hi*16);
      acc[s] = wmma_bf(a, bb, acc[s]);
    }
  }
  __syncthreads();   // staging no longer needed; reuse smem for G

  // scatter accumulators to G[o][n] (C/D layout: VGPR j -> M=j+8*hi, N=lane&15)
  #pragma unroll
  for(int s=0;s<4;s++)
    #pragma unroll
    for(int j=0;j<8;j++)
      G[(size_t)(wv*16 + j + hi*8)*GS + (s*16 + l)] = acc[s][j];
  __syncthreads();

  // gated activation: act = tanh(g[:64]) * sigmoid(g[64:])
  for(int idx = tid; idx < 64*64; idx += 256){
    int o = idx & 63, n = idx >> 6;
    float tg = tanhf(G[(size_t)o*GS + n]);
    float sg = 1.f / (1.f + expf(-G[(size_t)(o+64)*GS + n]));
    ACT[(size_t)n*AS_ + o] = (__bf16)(tg*sg);
  }
  __syncthreads();

  // ---- res/skip updates: waves 0..3 -> Wres, waves 4..7 -> Wskp
  const __bf16* Wm = (wv < 4) ? Wr : Wsk;
  int m2 = wv & 3;
  int row2 = m2*16 + l;
  v8f a4[4]; for(int s=0;s<4;s++) a4[s] = v8f_zero();
  #pragma unroll
  for(int kc=0;kc<2;kc++){
    v16bf a = ldA16(Wm + (size_t)row2*64 + kc*32 + hi*16);
    #pragma unroll
    for(int s=0;s<4;s++){
      v16bf bb = ldC16(ACT + (size_t)(s*16+l)*AS_ + kc*32 + hi*16);
      a4[s] = wmma_bf(a, bb, a4[s]);
    }
  }
  #pragma unroll
  for(int s=0;s<4;s++)
    #pragma unroll
    for(int j=0;j<8;j++){
      int o = m2*16 + j + hi*8;
      int n = s*16 + l;
      int gw = w0 + n;
      if(gw < WTOT){
        size_t idx = ((size_t)(bz*HTOT+h)*WTOT + gw)*64 + o;
        if(wv < 4) resOut[idx] = (__bf16)((float)resIn[idx] + a4[s][j]);
        else       skp[idx]    = (__bf16)((firstLayer ? 0.f : (float)skp[idx]) + a4[s][j]);
      }
    }
}

// ---------------------------------------------------------------- post net + flow update (WMMA)
__global__ __launch_bounds__(256) void k_post(
    const __bf16* __restrict__ skp, const __bf16* __restrict__ P1,
    const __bf16* __restrict__ P2, float* __restrict__ Xc,
    float* __restrict__ Gm, float* __restrict__ Gl, HPerm P, int flow0)
{
  __shared__ __align__(32) __bf16 SK[64*AS_];
  __shared__ __align__(32) __bf16 H1[64*AS_];
  int bz = blockIdx.z, h = blockIdx.y, w0 = blockIdx.x*64;
  int tid = threadIdx.x, lane = tid & 31, wv = tid >> 5;
  int l = lane & 15, hi = lane >> 4;

  for(int idx = tid; idx < 64*64; idx += 256){
    int n = idx >> 6, o = idx & 63;
    int gw = w0 + n;
    float v = (gw < WTOT) ? (float)skp[((size_t)(bz*HTOT+h)*WTOT + gw)*64 + o] : 0.f;
    SK[(size_t)n*AS_ + o] = (__bf16)fmaxf(v, 0.f);
  }
  __syncthreads();

  // h1 = relu(P1 @ relu(skp)) : M=64, N=64, K=64; 16 (m,ns) tiles over 8 waves
  int m = wv & 3;
  int ns0 = (wv >> 2) * 2;
  v8f acc[2]; acc[0] = v8f_zero(); acc[1] = v8f_zero();
  #pragma unroll
  for(int kc=0;kc<2;kc++){
    v16bf a = ldA16(P1 + (size_t)(m*16+l)*64 + kc*32 + hi*16);
    #pragma unroll
    for(int s=0;s<2;s++){
      v16bf bb = ldC16(SK + (size_t)((ns0+s)*16 + l)*AS_ + kc*32 + hi*16);
      acc[s] = wmma_bf(a, bb, acc[s]);
    }
  }
  #pragma unroll
  for(int s=0;s<2;s++)
    #pragma unroll
    for(int j=0;j<8;j++){
      int o = m*16 + j + hi*8;
      int n = (ns0+s)*16 + l;
      H1[(size_t)n*AS_ + o] = (__bf16)fmaxf(acc[s][j], 0.f);
    }
  __syncthreads();

  // out2 = P2pad(16x64) @ h1 ; rows 0/1 = mean/logvar
  if(wv < 4){
    int ns = wv;
    v8f a2 = v8f_zero();
    #pragma unroll
    for(int kc=0;kc<2;kc++){
      v16bf a = ldA16(P2 + (size_t)l*64 + kc*32 + hi*16);
      v16bf bb = ldC16(H1 + (size_t)(ns*16 + l)*AS_ + kc*32 + hi*16);
      a2 = wmma_bf(a, bb, a2);
    }
    if(hi == 0){ // M=0 (mean) in VGPR0, M=1 (logvar) in VGPR1, lanes 0-15
      int n = ns*16 + l;
      int gw = w0 + n;
      if(gw < WTOT){
        float mean = a2[0];
        float lv   = fminf(a2[1], 10.f);
        float e    = expf(lv);
        size_t fidx = (size_t)(bz*HTOT + h)*WTOT + gw;          // flow-space index
        size_t cidx = (size_t)(bz*HTOT + P.h[h])*WTOT + gw;     // canonical x index
        float xv = Xc[cidx];
        Xc[cidx] = e*xv + mean;
        if(flow0){ Gm[fidx] = mean;              Gl[fidx] = lv;       }
        else     { Gm[fidx] = Gm[fidx]*e + mean; Gl[fidx] = Gl[fidx] + lv; }
      }
    }
  }
}

// ---------------------------------------------------------------- final write-out
__global__ void k_final(const float* __restrict__ Xc, const float* __restrict__ Gm,
                        const float* __restrict__ Gl, float* __restrict__ out, HPerm P){
  int idx = blockIdx.x*256 + threadIdx.x;
  if(idx >= NPIX) return;
  int w = idx % WTOT;
  int h = (idx / WTOT) % HTOT;
  int b = idx / (HTOT*WTOT);
  out[idx]           = Xc[(size_t)(b*HTOT + P.h[h])*WTOT + w];
  out[NPIX + idx]    = Gm[idx];
  out[2*NPIX + idx]  = Gl[idx];
}

// ---------------------------------------------------------------- host
extern "C" void kernel_launch(void* const* d_in, const int* in_sizes, int n_in,
                              void* d_out, int out_size, void* d_ws, size_t ws_size,
                              hipStream_t stream) {
  (void)in_sizes; (void)n_in; (void)out_size; (void)ws_size;
  const float* x_in   = (const float*)d_in[0];
  const float* c_in   = (const float*)d_in[1];
  const float* firstw = (const float*)d_in[2];
  const float* initw  = (const float*)d_in[3];
  const float* cdtw   = (const float*)d_in[4];
  const float* resw   = (const float*)d_in[5];
  const float* skpw   = (const float*)d_in[6];
  const float* p1w    = (const float*)d_in[7];
  const float* p2w    = (const float*)d_in[8];

  char* ws = (char*)d_ws;
  size_t off = 0;
  auto take = [&](size_t bytes) -> char* {
    char* p = ws + off;
    off = (off + bytes + 255) & ~(size_t)255;
    return p;
  };
  __bf16* Wconv = (__bf16*)take((size_t)64*128*576*2);
  __bf16* Wcdt  = (__bf16*)take((size_t)64*128*96*2);
  __bf16* Wres  = (__bf16*)take((size_t)64*64*64*2);
  __bf16* Wskp  = (__bf16*)take((size_t)64*64*64*2);
  __bf16* Wp1   = (__bf16*)take((size_t)8*64*64*2);
  __bf16* Wp2   = (__bf16*)take((size_t)8*16*64*2);
  __bf16* Cb    = (__bf16*)take((size_t)NPIX*96*2);
  __bf16* ResA  = (__bf16*)take((size_t)NPIX*64*2);
  __bf16* ResB  = (__bf16*)take((size_t)NPIX*64*2);
  __bf16* Skp   = (__bf16*)take((size_t)NPIX*64*2);
  float*  Xc    = (float*) take((size_t)NPIX*4);
  float*  Gm    = (float*) take((size_t)NPIX*4);
  float*  Gl    = (float*) take((size_t)NPIX*4);

  // weight / input preparation
  k_prep_wconv<<<(64*128*576 + 255)/256, 256, 0, stream>>>(initw, Wconv);
  k_prep_wcdt <<<(64*128*96  + 255)/256, 256, 0, stream>>>(cdtw,  Wcdt);
  k_cvt<<<(64*64*64 + 255)/256, 256, 0, stream>>>(resw, Wres, 64*64*64);
  k_cvt<<<(64*64*64 + 255)/256, 256, 0, stream>>>(skpw, Wskp, 64*64*64);
  k_cvt<<<(8*64*64 + 255)/256, 256, 0, stream>>>(p1w, Wp1, 8*64*64);
  k_prep_wp2<<<(8*16*64 + 255)/256, 256, 0, stream>>>(p2w, Wp2);
  k_prep_c<<<(NPIX + 255)/256, 256, 0, stream>>>(c_in, Cb);
  k_cpy<<<(NPIX + 255)/256, 256, 0, stream>>>(x_in, Xc, NPIX);

  // cumulative H permutations: Q[0]=identity; Q[i+1] = Q[i] o flip_i
  HPerm Q[9];
  for(int h=0; h<16; h++) Q[0].h[h] = h;
  for(int i=0; i<8; i++)
    for(int h=0; h<16; h++){
      int fh = (i < 4) ? (15 - h) : ((h < 8) ? (7 - h) : (23 - h));
      Q[i+1].h[h] = Q[i].h[fh];
    }

  dim3 lgrid(NWTILE, HTOT, BTOT);
  for(int f=0; f<8; f++){
    k_first<<<(NPIX + 255)/256, 256, 0, stream>>>(Xc, firstw + (size_t)f*128, ResA, Q[f]);
    for(int l=0; l<8; l++){
      int d = 1 << (l & 3);
      const __bf16* rin = (l & 1) ? ResB : ResA;
      __bf16*       rout= (l & 1) ? ResA : ResB;
      size_t fl = (size_t)f*8 + l;
      k_layer<<<lgrid, 256, 0, stream>>>(rin, rout, Skp, Cb,
          Wconv + fl*128*576, Wcdt + fl*128*96,
          Wres  + fl*64*64,   Wskp + fl*64*64,
          d, (l == 0) ? 1 : 0, Q[f]);
    }
    k_post<<<lgrid, 256, 0, stream>>>(Skp, Wp1 + (size_t)f*64*64, Wp2 + (size_t)f*16*64,
                                      Xc, Gm, Gl, Q[f], (f == 0) ? 1 : 0);
  }
  k_final<<<(NPIX + 255)/256, 256, 0, stream>>>(Xc, Gm, Gl, (float*)d_out, Q[8]);
}
